// MaxCutScoreNet_37486474559590
// MI455X (gfx1250) — matrix operand order repaired and checked by reading, not compiled, not run on hardware
//
#include <hip/hip_runtime.h>
#include <hip/hip_bf16.h>
#include <math.h>

typedef __attribute__((ext_vector_type(16))) _Float16 v16h;
typedef __attribute__((ext_vector_type(8)))  float    v8f;

#define DELTA_C 2.0f
#define F_IN 128
#define HSTRIDE 32    // padded row stride for intermediate h buffers

// ---------------------------------------------------------------- utilities

__global__ void k_zero(float* __restrict__ p, int n) {
    int i = blockIdx.x * blockDim.x + threadIdx.x;
    if (i < n) p[i] = 0.0f;
}

// ------------------------------------------------- degree + edge weights

__global__ void k_deg(const int* __restrict__ col, const float* __restrict__ ew,
                      float* __restrict__ deg, int E) {
    int i = blockIdx.x * blockDim.x + threadIdx.x;
    if (i < E) atomicAdd(&deg[col[i]], ew[i]);
}

__global__ void k_edgew(const int* __restrict__ row, const int* __restrict__ col,
                        const float* __restrict__ ew, const float* __restrict__ deg,
                        float* __restrict__ w, int E) {
    int i = blockIdx.x * blockDim.x + threadIdx.x;
    if (i >= E) return;
    float dr = deg[row[i]];
    float dc = deg[col[i]];
    float ir = dr > 0.0f ? rsqrtf(dr) : 0.0f;
    float ic = dc > 0.0f ? rsqrtf(dc) : 0.0f;
    w[i] = (-DELTA_C) * ir * ew[i] * ic;
}

// -------------------------------------- weight conversion to f16 B layout
// w16: column-major, column stride Kpad (mult of 32), cols padded to outPad
// (mult of 16) with zeros.  A lane's B fragment for chunk kc is the 16
// halves at w16[n*Kpad + kc + 8*(lane>=16) + {0..7, 16..23}].

__global__ void k_prep_w(const float* __restrict__ W, _Float16* __restrict__ w16,
                         int K, int outW, int Kpad, int outPad) {
    int idx = blockIdx.x * blockDim.x + threadIdx.x;
    int total = outPad * Kpad;
    if (idx >= total) return;
    int n = idx / Kpad;
    int k = idx - n * Kpad;
    float v = (n < outW && k < K) ? W[k * outW + n] : 0.0f;  // W is (K,outW) row-major
    w16[idx] = (_Float16)v;
}

// ----------------------------------------------------- WMMA dense GEMM
// hw[N,OUTW] = h[N, K(padded to KPAD, zero tail)] @ W ; agg gets the same
// values (self-loop term, weight 1.0) so no separate copy pass is needed.
// One wave handles MT row tiles of one 16-col tile; KPAD/OUTW compile-time:
// chunk loop fully unrolls, B fragments stay in registers, C stores use
// immediate offsets off one base address.

template <int KPAD, int OUTW, int MT>
__global__ __launch_bounds__(128) void k_gemm(
    const float* __restrict__ h, int hs,          // h row stride (mult of 32)
    const _Float16* __restrict__ w16,
    float* __restrict__ hw, float* __restrict__ agg,
    int nRows, int rowGroups)
{
    constexpr int NC = KPAD / 32;
    constexpr int CT = (OUTW + 15) / 16;          // col tiles
    int wave = threadIdx.x >> 5;
    int lane = threadIdx.x & 31;
    int g = blockIdx.x * 4 + wave;
    if (g >= rowGroups * CT) return;              // whole-wave exit

    int rg   = g % rowGroups;
    int ct   = g / rowGroups;
    int col0 = ct * 16;
    int half = lane >> 4;        // 0: lanes 0-15, 1: lanes 16-31
    int mloc = lane & 15;
    int rowTiles = (nRows + 15) >> 4;

    // B fragments (all K chunks), loaded once per wave
    v16h bfrag[NC];
    const _Float16* wbase = w16 + (col0 + mloc) * KPAD + half * 8;
#pragma unroll
    for (int c = 0; c < NC; ++c) {
#pragma unroll
        for (int j = 0; j < 16; ++j) bfrag[c][j] = wbase[c * 32 + (j < 8 ? j : j + 8)];
    }

#pragma unroll
    for (int t = 0; t < MT; ++t) {
        int rt = rg * MT + t;                      // wave-uniform
        if (rt >= rowTiles) break;
        int row0 = rt * 16;
        const float* hrow = h + (size_t)(row0 + mloc) * hs;

        v8f acc = {};
#pragma unroll
        for (int kc = 0; kc < KPAD; kc += 32) {
            // A (16-bit 16x32 layout): lanes 0-15: K kc+0..7 / kc+16..23,
            //                          lanes 16-31: K kc+8..15 / kc+24..31
            v16h a;
            const float* pLo = hrow + kc + half * 8;
            const float* pHi = hrow + kc + 16 + half * 8;
            float4 f0 = ((const float4*)pLo)[0];
            float4 f1 = ((const float4*)pLo)[1];
            float4 f2 = ((const float4*)pHi)[0];
            float4 f3 = ((const float4*)pHi)[1];
            a[0]  = (_Float16)f0.x; a[1]  = (_Float16)f0.y;
            a[2]  = (_Float16)f0.z; a[3]  = (_Float16)f0.w;
            a[4]  = (_Float16)f1.x; a[5]  = (_Float16)f1.y;
            a[6]  = (_Float16)f1.z; a[7]  = (_Float16)f1.w;
            a[8]  = (_Float16)f2.x; a[9]  = (_Float16)f2.y;
            a[10] = (_Float16)f2.z; a[11] = (_Float16)f2.w;
            a[12] = (_Float16)f3.x; a[13] = (_Float16)f3.y;
            a[14] = (_Float16)f3.z; a[15] = (_Float16)f3.w;

            acc = __builtin_amdgcn_wmma_f32_16x16x32_f16(
                    false, a, false, bfrag[kc / 32], (short)0, acc, false, false);
        }

        // C layout: VGPR r -> M = row0 + r + 8*half; N = col0 + (lane&15)
        int n = col0 + mloc;
        if (OUTW >= 16 || mloc < OUTW) {           // compile-time for OUTW>=16
            size_t base = (size_t)(row0 + 8 * half) * OUTW + n;
#pragma unroll
            for (int r = 0; r < 8; ++r) {
                hw[base + r * OUTW]  = acc[r];
                agg[base + r * OUTW] = acc[r];     // self-loop init
            }
        }
    }
}

// ------------------------------------------------ sparse scatter (edges)
// One thread per (edge, 4 features): float4 gather + 4 atomic f32 adds.

__global__ void k_scatter4(const int* __restrict__ row, const int* __restrict__ col,
                           const float* __restrict__ w, const float* __restrict__ hw,
                           float* __restrict__ agg, int E, int F) {
    int q = F >> 2;
    int idx = blockIdx.x * blockDim.x + threadIdx.x;
    if (idx >= E * q) return;
    int e = idx / q;
    int v = (idx - e * q) << 2;
    int r = row[e];
    int c = col[e];
    float we = w[e];
    float4 hv = *(const float4*)(hw + (size_t)r * F + v);
    float* dst = agg + (size_t)c * F + v;
    atomicAdd(dst + 0, we * hv.x);
    atomicAdd(dst + 1, we * hv.y);
    atomicAdd(dst + 2, we * hv.z);
    atomicAdd(dst + 3, we * hv.w);
}

// bias + tanh into padded h buffer (stride FS, zero tail columns)
__global__ void k_bias_tanh_pad(const float* __restrict__ agg, const float* __restrict__ b,
                                float* __restrict__ h, int n, int F, int FS) {
    int idx = blockIdx.x * blockDim.x + threadIdx.x;
    if (idx >= n * FS) return;
    int i = idx / FS;
    int f = idx - i * FS;
    h[idx] = (f < F) ? tanhf(agg[(size_t)i * F + f] + b[f]) : 0.0f;
}

// --------------------------------------------------------- MLP head
// h(8, stride HSTRIDE) -> relu 16 -> relu 16 -> tanh scalar

__global__ void k_mlp(const float* __restrict__ h,
                      const float* __restrict__ W0, const float* __restrict__ b0,
                      const float* __restrict__ W1, const float* __restrict__ b1,
                      const float* __restrict__ fw, const float* __restrict__ fb,
                      float* __restrict__ out, int n) {
    __shared__ float sW0[8 * 16];
    __shared__ float sW1[16 * 16];
    __shared__ float sb0[16], sb1[16], sfw[16], sfb[1];
    for (int i = threadIdx.x; i < 128; i += blockDim.x) sW0[i] = W0[i];
    for (int i = threadIdx.x; i < 256; i += blockDim.x) sW1[i] = W1[i];
    if (threadIdx.x < 16) {
        sb0[threadIdx.x] = b0[threadIdx.x];
        sb1[threadIdx.x] = b1[threadIdx.x];
        sfw[threadIdx.x] = fw[threadIdx.x];
    }
    if (threadIdx.x == 0) sfb[0] = fb[0];
    __syncthreads();

    int i = blockIdx.x * blockDim.x + threadIdx.x;
    if (i >= n) return;

    float hin[8];
#pragma unroll
    for (int k = 0; k < 8; ++k) hin[k] = h[(size_t)i * HSTRIDE + k];

    float t1[16];
#pragma unroll
    for (int j = 0; j < 16; ++j) {
        float s = sb0[j];
#pragma unroll
        for (int k = 0; k < 8; ++k) s += hin[k] * sW0[k * 16 + j];
        t1[j] = s > 0.0f ? s : 0.0f;
    }
    float t2[16];
#pragma unroll
    for (int j = 0; j < 16; ++j) {
        float s = sb1[j];
#pragma unroll
        for (int k = 0; k < 16; ++k) s += t1[k] * sW1[k * 16 + j];
        t2[j] = s > 0.0f ? s : 0.0f;
    }
    float s = sfb[0];
#pragma unroll
    for (int j = 0; j < 16; ++j) s += t2[j] * sfw[j];
    out[i] = tanhf(s);
}

// ---------------------------------------------------------------- launch

extern "C" void kernel_launch(void* const* d_in, const int* in_sizes, int n_in,
                              void* d_out, int out_size, void* d_ws, size_t ws_size,
                              hipStream_t stream) {
    const float* x    = (const float*)d_in[0];
    const int*   ei   = (const int*)d_in[1];          // (2, E) int32
    const float* ew   = (const float*)d_in[2];
    const int    N    = in_sizes[0] / F_IN;
    const int    E    = in_sizes[2];
    const int*   rows = ei;
    const int*   cols = ei + E;

    const float* gcn_w[12]; const float* gcn_b[12];
    for (int l = 0; l < 12; ++l) { gcn_w[l] = (const float*)d_in[3 + l];
                                   gcn_b[l] = (const float*)d_in[15 + l]; }
    const float* mlp_w0 = (const float*)d_in[27];
    const float* mlp_w1 = (const float*)d_in[28];
    const float* mlp_b0 = (const float*)d_in[29];
    const float* mlp_b1 = (const float*)d_in[30];
    const float* fin_w  = (const float*)d_in[31];
    const float* fin_b  = (const float*)d_in[32];

    static const int gcnK[12]   = {128,32,32,32,32,16,16,16,16,8,8,8};
    static const int gcnOut[12] = {32,32,32,32,16,16,16,16,8,8,8,8};

    // ---- workspace carve (256B aligned)
    char* ws = (char*)d_ws;
    size_t off = 0;
    auto carve = [&](size_t bytes) -> void* {
        void* p = ws + off;
        off = (off + bytes + 255) & ~(size_t)255;
        return p;
    };
    float*    deg  = (float*)carve((size_t)N * 4);
    float*    wbuf = (float*)carve((size_t)E * 4);
    float*    hw   = (float*)carve((size_t)N * 32 * 4);
    float*    agg  = (float*)carve((size_t)N * 32 * 4);
    float*    hbuf = (float*)carve((size_t)N * HSTRIDE * 4);
    _Float16* w16[12];
    int Kpad[12], outPad[12];
    for (int l = 0; l < 12; ++l) {
        Kpad[l]   = ((gcnK[l] + 31) / 32) * 32;       // 128 or 32
        outPad[l] = ((gcnOut[l] + 15) / 16) * 16;     // 32 or 16
        w16[l] = (_Float16*)carve((size_t)Kpad[l] * outPad[l] * 2);
    }
    (void)ws_size; (void)n_in;

    const int TB = 256;
    auto blocks = [](int total, int tb) { return (total + tb - 1) / tb; };

    // ---- graph normalization
    k_zero<<<blocks(N, TB), TB, 0, stream>>>(deg, N);
    k_deg<<<blocks(E, TB), TB, 0, stream>>>(cols, ew, deg, E);
    k_edgew<<<blocks(E, TB), TB, 0, stream>>>(rows, cols, ew, deg, wbuf, E);

    // ---- weight conversion (tiny)
    for (int l = 0; l < 12; ++l) {
        int total = Kpad[l] * outPad[l];
        k_prep_w<<<blocks(total, TB), TB, 0, stream>>>(
            gcn_w[l], w16[l], gcnK[l], gcnOut[l], Kpad[l], outPad[l]);
    }

    // ---- 12 GCN layers
    const int rowTiles  = (N + 15) / 16;              // 6250
    const int MT        = 4;
    const int rowGroups = (rowTiles + MT - 1) / MT;   // 1563
    const float* h = x;
    int hs = F_IN;
    for (int l = 0; l < 12; ++l) {
        int outW = gcnOut[l];
        int colTiles = outPad[l] / 16;
        int groups = rowGroups * colTiles;
        int gb = blocks(groups, 4);
        if (l == 0) {
            k_gemm<128, 32, MT><<<gb, 128, 0, stream>>>(h, hs, w16[l], hw, agg, N, rowGroups);
        } else if (outW == 32) {
            k_gemm<32, 32, MT><<<gb, 128, 0, stream>>>(h, hs, w16[l], hw, agg, N, rowGroups);
        } else if (outW == 16) {
            k_gemm<32, 16, MT><<<gb, 128, 0, stream>>>(h, hs, w16[l], hw, agg, N, rowGroups);
        } else {
            k_gemm<32, 8, MT><<<gb, 128, 0, stream>>>(h, hs, w16[l], hw, agg, N, rowGroups);
        }
        k_scatter4<<<blocks(E * (outW / 4), TB), TB, 0, stream>>>(
            rows, cols, wbuf, hw, agg, E, outW);
        k_bias_tanh_pad<<<blocks(N * HSTRIDE, TB), TB, 0, stream>>>(
            agg, gcn_b[l], hbuf, N, outW, HSTRIDE);
        h = hbuf;
        hs = HSTRIDE;
    }

    // ---- MLP head + final score
    k_mlp<<<blocks(N, TB), TB, 0, stream>>>(
        h, mlp_w0, mlp_b0, mlp_w1, mlp_b1, fin_w, fin_b, (float*)d_out, N);
    (void)out_size;
}
